// IterativeFlowRefiner_25366076850704
// MI455X (gfx1250) — matrix operand order repaired and checked by reading, not compiled, not run on hardware
//
#include <hip/hip_runtime.h>
#include <hip/hip_bf16.h>
#include <math.h>

typedef __attribute__((ext_vector_type(16))) __bf16 v16bf;
typedef __attribute__((ext_vector_type(8)))  __bf16 v8bf;
typedef __attribute__((ext_vector_type(8)))  float  v8f;

#define MTOT  8192   // B*H*W
#define HH    64
#define WW    64
#define HW    4096

__device__ __forceinline__ float sigm_f(float x){ return 1.f/(1.f+expf(-x)); }
__device__ __forceinline__ float gelu_f(float x){ return 0.5f*x*(1.f+erff(x*0.7071067811865476f)); }
__device__ __forceinline__ float act_f(float v, int act){
  if (act==1) return fmaxf(v, 0.f);
  if (act==2) return gelu_f(v);
  return v;
}

// A fragment (16-bit A 16x32): row = lane%16,
// elems 0..15 -> K = k0 + (lane/16)*16 + j (matches B fill).
__device__ __forceinline__ v16bf load_afrag(const __bf16* arow, int k0, int lh){
  v16bf a = *(const v16bf*)(arow + k0 + lh*16);
  return a;
}

// =============================================================================
// WMMA GEMM (16x16 tile / wave) — used only for ragged-N heads (Npad 16/48).
// C[M x Ns](f32) = act( alpha * A[M x Ka](bf16 rm) * W[Ns x Ka](bf16 rm)^T + bias )
// =============================================================================
__global__ void wmma_gemm_k(const __bf16* __restrict__ A, const __bf16* __restrict__ W,
                            const float* __restrict__ bias, float* __restrict__ C,
                            int Ka, int Ns, float alpha, int act){
  const int lane = threadIdx.x & 31;
  const int lr   = lane & 15;
  const int lh   = lane >> 4;
  const long tileM = (long)blockIdx.y << 4;
  const long tileN = (long)blockIdx.x << 4;
  const __bf16* arow = A + (tileM + lr) * (long)Ka;
  const __bf16* wrow = W + (tileN + lr) * (long)Ka;
  v8f c;
#pragma unroll
  for (int i=0;i<8;i++) c[i]=0.f;
  for (int k0=0;k0<Ka;k0+=32){
    v16bf a = load_afrag(arow, k0, lh);
    v16bf b = *(const v16bf*)(wrow + k0 + lh*16);   // B 32x16: col=lane%16, K=lh*16+j
    c = __builtin_amdgcn_wmma_f32_16x16x32_bf16(false, a, false, b, (short)0, c, false, false);
  }
  const int outCol = (int)tileN + lr;
  const float bv = bias ? bias[outCol] : 0.f;
#pragma unroll
  for (int r=0;r<8;r++)
    C[(tileM + (long)r + lh*8)*(long)Ns + outCol] = act_f(c[r]*alpha + bv, act);
}

// =============================================================================
// WMMA GEMM (32x32 tile / wave, 2x2 register blocking): 4 WMMAs per K-step
// reuse each A/B fragment twice -> 1KB fragment traffic per WMMA (2x intensity).
// Requires M%32==0 and Ns%32==0.
// =============================================================================
__global__ void wmma_gemm2_k(const __bf16* __restrict__ A, const __bf16* __restrict__ W,
                             const float* __restrict__ bias, float* __restrict__ C,
                             int Ka, int Ns, float alpha, int act){
  const int lane = threadIdx.x & 31;
  const int lr   = lane & 15;
  const int lh   = lane >> 4;
  const long tileM = (long)blockIdx.y << 5;
  const long tileN = (long)blockIdx.x << 5;
  const __bf16* arow0 = A + (tileM + lr) * (long)Ka;
  const __bf16* arow1 = arow0 + 16*(long)Ka;
  const __bf16* wrow0 = W + (tileN + lr) * (long)Ka;
  const __bf16* wrow1 = wrow0 + 16*(long)Ka;
  v8f c00, c01, c10, c11;
#pragma unroll
  for (int i=0;i<8;i++){ c00[i]=0.f; c01[i]=0.f; c10[i]=0.f; c11[i]=0.f; }
  for (int k0=0;k0<Ka;k0+=32){
    v16bf a0 = load_afrag(arow0, k0, lh);
    v16bf a1 = load_afrag(arow1, k0, lh);
    v16bf b0 = *(const v16bf*)(wrow0 + k0 + lh*16);
    v16bf b1 = *(const v16bf*)(wrow1 + k0 + lh*16);
    c00 = __builtin_amdgcn_wmma_f32_16x16x32_bf16(false, a0, false, b0, (short)0, c00, false, false);
    c10 = __builtin_amdgcn_wmma_f32_16x16x32_bf16(false, a1, false, b0, (short)0, c10, false, false);
    c01 = __builtin_amdgcn_wmma_f32_16x16x32_bf16(false, a0, false, b1, (short)0, c01, false, false);
    c11 = __builtin_amdgcn_wmma_f32_16x16x32_bf16(false, a1, false, b1, (short)0, c11, false, false);
  }
  const int col0 = (int)tileN + lr;
  const int col1 = col0 + 16;
  const float bv0 = bias ? bias[col0] : 0.f;
  const float bv1 = bias ? bias[col1] : 0.f;
#pragma unroll
  for (int r=0;r<8;r++){
    long row0 = tileM + r + lh*8;
    long row1 = row0 + 16;
    C[row0*(long)Ns + col0] = act_f(c00[r]*alpha + bv0, act);
    C[row0*(long)Ns + col1] = act_f(c01[r]*alpha + bv1, act);
    C[row1*(long)Ns + col0] = act_f(c10[r]*alpha + bv0, act);
    C[row1*(long)Ns + col1] = act_f(c11[r]*alpha + bv1, act);
  }
}

// ---------------- weight / bias conversion (f32 OIHW -> padded bf16 [Npad,Kpad]) -
__global__ void conv_w_k(const float* __restrict__ src, __bf16* __restrict__ dst,
                         int N, int K, int Npad, int Kpad){
  long idx = (long)blockIdx.x*blockDim.x + threadIdx.x;
  long tot = (long)Npad*Kpad;
  if (idx >= tot) return;
  int n = (int)(idx / Kpad), k = (int)(idx % Kpad);
  float v = (n < N && k < K) ? src[(long)n*K + k] : 0.f;
  dst[idx] = (__bf16)v;
}
__global__ void conv_b_k(const float* __restrict__ src, float* __restrict__ dst, int N, int Npad){
  int n = blockIdx.x*blockDim.x + threadIdx.x;
  if (n >= Npad) return;
  dst[n] = (n < N) ? src[n] : 0.f;
}

// ---------------- im2col: NCHW f32 -> bf16 patch matrix [MTOT, Kpad] ------------
__global__ void im2col_nchw_k(const float* __restrict__ src, __bf16* __restrict__ dst,
                              int Cin, int KH, int KW, int padH, int padW, int Kpad){
  long idx = (long)blockIdx.x*blockDim.x + threadIdx.x;
  long tot = (long)MTOT*Kpad;
  if (idx >= tot) return;
  int m = (int)(idx / Kpad), k = (int)(idx % Kpad);
  int Kval = Cin*KH*KW;
  float v = 0.f;
  if (k < Kval){
    int ci = k/(KH*KW), rem = k%(KH*KW), kh = rem/KW, kw = rem%KW;
    int b = m >> 12, y = (m >> 6) & 63, x = m & 63;
    int yy = y + kh - padH, xx = x + kw - padW;
    if (yy>=0 && yy<HH && xx>=0 && xx<WW)
      v = src[(((long)b*Cin + ci)*HH + yy)*WW + xx];
  }
  dst[idx] = (__bf16)v;
}

// ---------------- im2col: pixel-major f32 [MTOT, Cs] -> bf16 [MTOT, Kpad] -------
__global__ void im2col_pm_k(const float* __restrict__ src, __bf16* __restrict__ dst,
                            int Cs, int Cin, int KH, int KW, int padH, int padW, int Kpad){
  long idx = (long)blockIdx.x*blockDim.x + threadIdx.x;
  long tot = (long)MTOT*Kpad;
  if (idx >= tot) return;
  int m = (int)(idx / Kpad), k = (int)(idx % Kpad);
  int Kval = Cin*KH*KW;
  float v = 0.f;
  if (k < Kval){
    int ci = k/(KH*KW), rem = k%(KH*KW), kh = rem/KW, kw = rem%KW;
    int b = m >> 12, y = (m >> 6) & 63, x = m & 63;
    int yy = y + kh - padH, xx = x + kw - padW;
    if (yy>=0 && yy<HH && xx>=0 && xx<WW)
      v = src[(((long)b*HH + yy)*WW + xx)*(long)Cs + ci];
  }
  dst[idx] = (__bf16)v;
}

// ---------------- l2norm over 128 channels, f32 -> bf16 ------------------------
__global__ void l2norm_k(const float* __restrict__ src, __bf16* __restrict__ dst){
  int m = blockIdx.x*blockDim.x + threadIdx.x;
  if (m >= MTOT) return;
  float s = 0.f;
  for (int c=0;c<128;c++){ float v = src[(long)m*128+c]; s += v*v; }
  float inv = 1.f / fmaxf(sqrtf(s), 1e-12f);
  for (int c=0;c<128;c++) dst[(long)m*128+c] = (__bf16)(src[(long)m*128+c]*inv);
}

// ---------------- 2x2 average pooling over last two dims -----------------------
__global__ void avgpool_k(const float* __restrict__ src, float* __restrict__ dst, int hh){
  int ww = hh, oh = hh>>1, ow = ww>>1;
  long idx = (long)blockIdx.x*blockDim.x + threadIdx.x;
  long tot = (long)MTOT*oh*ow;
  if (idx >= tot) return;
  int m = (int)(idx/(oh*ow)), r = (int)(idx%(oh*ow));
  int y = r/ow, x = r%ow;
  const float* s = src + (long)m*hh*ww;
  dst[idx] = 0.25f*(s[(2*y)*ww+2*x] + s[(2*y)*ww+2*x+1] + s[(2*y+1)*ww+2*x] + s[(2*y+1)*ww+2*x+1]);
}

// ---------------- correlation pyramid lookup -> bf16 [MTOT, 352] ----------------
__device__ __forceinline__ float bilin(const float* img, int hh, int ww, float px, float py){
  int x0 = (int)floorf(px), y0 = (int)floorf(py);
  float wx = px - (float)x0, wy = py - (float)y0;
  float acc = 0.f;
#pragma unroll
  for (int dy=0;dy<2;dy++)
#pragma unroll
    for (int dx=0;dx<2;dx++){
      int xi = x0+dx, yi = y0+dy;
      if (xi>=0 && xi<ww && yi>=0 && yi<hh){
        float wt = (dx ? wx : 1.f-wx)*(dy ? wy : 1.f-wy);
        acc += img[(long)yi*ww + xi]*wt;
      }
    }
  return acc;
}
__global__ void corr_lookup_k(const float* __restrict__ p0, const float* __restrict__ p1,
                              const float* __restrict__ p2, const float* __restrict__ p3,
                              const float* __restrict__ coords, __bf16* __restrict__ dst){
  long idx = (long)blockIdx.x*blockDim.x + threadIdx.x;
  long tot = (long)MTOT*352;
  if (idx >= tot) return;
  int m = (int)(idx/352), k = (int)(idx%352);
  float v = 0.f;
  if (k < 324){
    int l = k/81, p = k%81;
    // faithful to reference: ox comes from the 'ij' dy-grid, oy from the dx-grid
    float ox = (float)(p/9) - 4.f;
    float oy = (float)(p%9) - 4.f;
    float cx = coords[(long)m*2+0], cy = coords[(long)m*2+1];
    int hh = HH >> l, ww = WW >> l;
    float s = 1.f/(float)(1<<l);
    const float* base = (l==0 ? p0 : l==1 ? p1 : l==2 ? p2 : p3) + (long)m*hh*ww;
    v = bilin(base, hh, ww, cx*s + ox, cy*s + oy);
  }
  dst[idx] = (__bf16)v;
}

// ---------------- small elementwise helpers -------------------------------------
__global__ void cnet_split_k(const float* __restrict__ cnet, float* __restrict__ net,
                             float* __restrict__ inp){
  long idx = (long)blockIdx.x*blockDim.x + threadIdx.x;
  if (idx >= (long)MTOT*128) return;
  int m = (int)(idx/128), c = (int)(idx%128);
  net[idx] = tanhf(cnet[(long)m*256 + c]);
  inp[idx] = fmaxf(cnet[(long)m*256 + 128 + c], 0.f);
}
__global__ void coords_init_k(const float* __restrict__ pf2, float* __restrict__ coords){
  int m = blockIdx.x*blockDim.x + threadIdx.x;
  if (m >= MTOT) return;
  float bx = (float)(m & 63), by = (float)((m >> 6) & 63);
  coords[(long)m*2+0] = bx + pf2[(long)m*16+0];
  coords[(long)m*2+1] = by + pf2[(long)m*16+1];
}
__global__ void flow_k(const float* __restrict__ coords, float* __restrict__ flow){
  int m = blockIdx.x*blockDim.x + threadIdx.x;
  if (m >= MTOT) return;
  flow[(long)m*2+0] = coords[(long)m*2+0] - (float)(m & 63);
  flow[(long)m*2+1] = coords[(long)m*2+1] - (float)((m >> 6) & 63);
}
__global__ void coords_upd_k(float* __restrict__ coords, const float* __restrict__ d){
  int m = blockIdx.x*blockDim.x + threadIdx.x;
  if (m >= MTOT) return;
  coords[(long)m*2+0] += d[(long)m*16+0];
  coords[(long)m*2+1] += d[(long)m*16+1];
}
__global__ void concat2_k(const float* __restrict__ A, int Csa, int Ca,
                          const float* __restrict__ Bp, int Csb, int Cb,
                          float* __restrict__ dst){
  int Co = Ca + Cb;
  long idx = (long)blockIdx.x*blockDim.x + threadIdx.x;
  if (idx >= (long)MTOT*Co) return;
  int m = (int)(idx/Co), c = (int)(idx%Co);
  dst[(long)m*Co + c] = (c < Ca) ? A[(long)m*Csa + c] : Bp[(long)m*Csb + (c - Ca)];
}
__global__ void rqcat_k(const float* __restrict__ rlin, const float* __restrict__ net,
                        const float* __restrict__ x, float* __restrict__ dst){
  long idx = (long)blockIdx.x*blockDim.x + threadIdx.x;
  if (idx >= (long)MTOT*384) return;
  int m = (int)(idx/384), c = (int)(idx%384);
  dst[idx] = (c < 128) ? sigm_f(rlin[(long)m*128 + c])*net[(long)m*128 + c]
                       : x[(long)m*256 + (c - 128)];
}
__global__ void gru_upd_k(float* __restrict__ net, const float* __restrict__ zlin,
                          const float* __restrict__ qlin){
  long idx = (long)blockIdx.x*blockDim.x + threadIdx.x;
  if (idx >= (long)MTOT*128) return;
  float z = sigm_f(zlin[idx]);
  float q = tanhf(qlin[idx]);
  net[idx] = (1.f - z)*net[idx] + z*q;
}

// ---------------- convex upsample (2x) ------------------------------------------
__global__ void upsample_k(const float* __restrict__ flow, const float* __restrict__ mask,
                           float* __restrict__ out){
  int idx = blockIdx.x*blockDim.x + threadIdx.x;
  if (idx >= 2*2*128*128) return;
  int X = idx & 127, Y = (idx >> 7) & 127, ch = (idx >> 14) & 1, n = idx >> 15;
  int x = X >> 1, c2 = X & 1, y = Y >> 1, r2 = Y & 1;
  int m = (n*64 + y)*64 + x;
  float w[9], mx = -1e30f;
#pragma unroll
  for (int k=0;k<9;k++){
    w[k] = 0.25f*mask[(long)m*48 + k*4 + r2*2 + c2];
    mx = fmaxf(mx, w[k]);
  }
  float sum = 0.f;
#pragma unroll
  for (int k=0;k<9;k++){ w[k] = expf(w[k]-mx); sum += w[k]; }
  float inv = 1.f/sum, acc = 0.f;
#pragma unroll
  for (int k=0;k<9;k++){
    int yy = y + k/3 - 1, xx = x + k%3 - 1;
    float nb = 0.f;
    if (yy>=0 && yy<HH && xx>=0 && xx<WW)
      nb = 2.f*flow[((long)(n*64 + yy)*64 + xx)*2 + ch];
    acc += w[k]*inv*nb;
  }
  out[((long)(n*2 + ch)*128 + Y)*128 + X] = acc;
}

// =============================================================================
// Host orchestration
// =============================================================================
enum {
  IN_FEAT1=0, IN_FEAT2, IN_CTX, IN_PRIOR,
  P_FP_W, P_FP_B, P_CE1_W, P_CE1_B, P_CE2_W, P_CE2_B, P_PF1_W, P_PF1_B,
  P_PF2_W, P_PF2_B, P_MC1_W, P_MC1_B, P_MC2_W, P_MC2_B, P_MF1_W, P_MF1_B,
  P_MF2_W, P_MF2_B, P_MO_W, P_MO_B,
  P_GZ1_W, P_GZ1_B, P_GR1_W, P_GR1_B, P_GQ1_W, P_GQ1_B,
  P_GZ2_W, P_GZ2_B, P_GR2_W, P_GR2_B, P_GQ2_W, P_GQ2_B,
  P_FH1_W, P_FH1_B, P_FH2_W, P_FH2_B, P_MH1_W, P_MH1_B, P_MH2_W, P_MH2_B
};
enum { W_FP=0, W_CE1, W_CE2, W_PF1, W_PF2, W_MC1, W_MC2, W_MF1, W_MF2, W_MO,
       W_GZ1, W_GR1, W_GQ1, W_GZ2, W_GR2, W_GQ2, W_FH1, W_FH2, W_MH1, W_MH2, W_COUNT };

struct WDescT { int wi, bi, N, K; };
static const WDescT g_wtab[W_COUNT] = {
  {P_FP_W ,P_FP_B ,128, 256}, {P_CE1_W,P_CE1_B,256,2304}, {P_CE2_W,P_CE2_B,256,2304},
  {P_PF1_W,P_PF1_B, 64, 576}, {P_PF2_W,P_PF2_B,  2, 576}, {P_MC1_W,P_MC1_B,192, 324},
  {P_MC2_W,P_MC2_B,128,1728}, {P_MF1_W,P_MF1_B, 64,  98}, {P_MF2_W,P_MF2_B, 32, 576},
  {P_MO_W ,P_MO_B ,126,1440},
  {P_GZ1_W,P_GZ1_B,128,1920}, {P_GR1_W,P_GR1_B,128,1920}, {P_GQ1_W,P_GQ1_B,128,1920},
  {P_GZ2_W,P_GZ2_B,128,1920}, {P_GR2_W,P_GR2_B,128,1920}, {P_GQ2_W,P_GQ2_B,128,1920},
  {P_FH1_W,P_FH1_B,256,1152}, {P_FH2_W,P_FH2_B,  2,2304}, {P_MH1_W,P_MH1_B,256,1152},
  {P_MH2_W,P_MH2_B, 36, 256}
};
static inline int padN(int n){ return (n + 15) & ~15; }
static inline int padK(int k){ return (k + 31) & ~31; }
static inline unsigned gbl(long n){ return (unsigned)((n + 255)/256); }

extern "C" void kernel_launch(void* const* d_in, const int* in_sizes, int n_in,
                              void* d_out, int out_size, void* d_ws, size_t ws_size,
                              hipStream_t stream) {
  (void)in_sizes; (void)n_in; (void)out_size; (void)ws_size;
  char* ws = (char*)d_ws;

  // ---- workspace layout (deterministic every call) ----
  size_t off = 0;
  auto alloc = [&](size_t bytes)->size_t{
    size_t r = off; off += (bytes + 255) & ~(size_t)255; return r; };

  size_t woff[W_COUNT], boff[W_COUNT];
  for (int i=0;i<W_COUNT;i++){
    int Np = padN(g_wtab[i].N), Kp = padK(g_wtab[i].K);
    woff[i] = alloc((size_t)Np*Kp*sizeof(__bf16));
    boff[i] = alloc((size_t)Np*sizeof(float));
  }
  size_t o_bufA  = alloc((size_t)MTOT*2304*sizeof(__bf16));   // shared im2col / lookup buffer
  size_t o_fm1   = alloc((size_t)MTOT*128*sizeof(__bf16));
  size_t o_fm2   = alloc((size_t)MTOT*128*sizeof(__bf16));
  size_t o_pyr0  = alloc((size_t)MTOT*4096*sizeof(float));
  size_t o_pyr1  = alloc((size_t)MTOT*1024*sizeof(float));
  size_t o_pyr2  = alloc((size_t)MTOT*256*sizeof(float));
  size_t o_pyr3  = alloc((size_t)MTOT*64*sizeof(float));
  size_t o_t256a = alloc((size_t)MTOT*256*sizeof(float));
  size_t o_t256b = alloc((size_t)MTOT*256*sizeof(float));
  size_t o_net   = alloc((size_t)MTOT*128*sizeof(float));
  size_t o_inp   = alloc((size_t)MTOT*128*sizeof(float));
  size_t o_zlin  = alloc((size_t)MTOT*128*sizeof(float));
  size_t o_rlin  = alloc((size_t)MTOT*128*sizeof(float));
  size_t o_qlin  = alloc((size_t)MTOT*128*sizeof(float));
  size_t o_cor1  = alloc((size_t)MTOT*192*sizeof(float));
  size_t o_cor2  = alloc((size_t)MTOT*128*sizeof(float));
  size_t o_flo1  = alloc((size_t)MTOT*64*sizeof(float));
  size_t o_flo2  = alloc((size_t)MTOT*32*sizeof(float));
  size_t o_cf    = alloc((size_t)MTOT*160*sizeof(float));
  size_t o_mo    = alloc((size_t)MTOT*128*sizeof(float));
  size_t o_mot   = alloc((size_t)MTOT*128*sizeof(float));
  size_t o_x     = alloc((size_t)MTOT*256*sizeof(float));
  size_t o_hx    = alloc((size_t)MTOT*384*sizeof(float));
  size_t o_qin   = alloc((size_t)MTOT*384*sizeof(float));
  size_t o_crd   = alloc((size_t)MTOT*2*sizeof(float));
  size_t o_flow  = alloc((size_t)MTOT*2*sizeof(float));
  size_t o_t16   = alloc((size_t)MTOT*16*sizeof(float));
  size_t o_mask  = alloc((size_t)MTOT*48*sizeof(float));
  size_t o_pf1   = alloc((size_t)MTOT*64*sizeof(float));

  auto F  = [&](size_t o)->float*  { return (float*)(ws + o); };
  auto BH = [&](size_t o)->__bf16* { return (__bf16*)(ws + o); };
  auto IN = [&](int i)->const float*{ return (const float*)d_in[i]; };
  auto WGT = [&](int i)->const __bf16*{ return (const __bf16*)(ws + woff[i]); };
  auto BIA = [&](int i)->const float*{ return (const float*)(ws + boff[i]); };

  auto gemm = [&](const __bf16* A, int wid, float* C, float alpha, int act){
    int Np = padN(g_wtab[wid].N), Kp = padK(g_wtab[wid].K);
    if ((Np & 31) == 0){
      dim3 g(Np/32, MTOT/32);
      wmma_gemm2_k<<<g, 32, 0, stream>>>(A, WGT(wid), BIA(wid), C, Kp, Np, alpha, act);
    } else {
      dim3 g(Np/16, MTOT/16);
      wmma_gemm_k<<<g, 32, 0, stream>>>(A, WGT(wid), BIA(wid), C, Kp, Np, alpha, act);
    }
  };
  auto ic_nchw = [&](const float* src, int Cin, int KH, int KW, int pH, int pW, int Kp){
    im2col_nchw_k<<<gbl((long)MTOT*Kp), 256, 0, stream>>>(src, BH(o_bufA), Cin, KH, KW, pH, pW, Kp);
  };
  auto ic_pm = [&](const float* src, int Cs, int Cin, int KH, int KW, int pH, int pW, int Kp){
    im2col_pm_k<<<gbl((long)MTOT*Kp), 256, 0, stream>>>(src, BH(o_bufA), Cs, Cin, KH, KW, pH, pW, Kp);
  };

  // ---- weight / bias conversion ----
  for (int i=0;i<W_COUNT;i++){
    int Np = padN(g_wtab[i].N), Kp = padK(g_wtab[i].K);
    conv_w_k<<<gbl((long)Np*Kp), 256, 0, stream>>>(IN(g_wtab[i].wi), BH(woff[i]),
                                                   g_wtab[i].N, g_wtab[i].K, Np, Kp);
    conv_b_k<<<gbl(Np), 256, 0, stream>>>(IN(g_wtab[i].bi), F(boff[i]), g_wtab[i].N, Np);
  }

  // ---- feature projection + l2norm (uses o_zlin as f32 temp) ----
  ic_nchw(IN(IN_FEAT1), 256, 1, 1, 0, 0, 256);
  gemm(BH(o_bufA), W_FP, F(o_zlin), 1.f, 0);
  l2norm_k<<<gbl(MTOT), 256, 0, stream>>>(F(o_zlin), BH(o_fm1));
  ic_nchw(IN(IN_FEAT2), 256, 1, 1, 0, 0, 256);
  gemm(BH(o_bufA), W_FP, F(o_zlin), 1.f, 0);
  l2norm_k<<<gbl(MTOT), 256, 0, stream>>>(F(o_zlin), BH(o_fm2));

  // ---- all-pairs correlation: per batch 4096x4096x128 WMMA GEMM ----
  const float cscale = 0.08838834764831845f; // 1/sqrt(128)
  for (int b=0;b<2;b++){
    dim3 g(4096/32, 4096/32);
    wmma_gemm2_k<<<g, 32, 0, stream>>>(BH(o_fm1) + (size_t)b*4096*128,
                                       BH(o_fm2) + (size_t)b*4096*128,
                                       nullptr,
                                       F(o_pyr0) + (size_t)b*4096*4096,
                                       128, 4096, cscale, 0);
  }
  avgpool_k<<<gbl((long)MTOT*32*32), 256, 0, stream>>>(F(o_pyr0), F(o_pyr1), 64);
  avgpool_k<<<gbl((long)MTOT*16*16), 256, 0, stream>>>(F(o_pyr1), F(o_pyr2), 32);
  avgpool_k<<<gbl((long)MTOT*8*8),   256, 0, stream>>>(F(o_pyr2), F(o_pyr3), 16);

  // ---- context encoder ----
  ic_nchw(IN(IN_CTX), 256, 3, 3, 1, 1, 2304);
  gemm(BH(o_bufA), W_CE1, F(o_t256a), 1.f, 2);             // gelu
  ic_pm(F(o_t256a), 256, 256, 3, 3, 1, 1, 2304);
  gemm(BH(o_bufA), W_CE2, F(o_t256b), 1.f, 0);
  cnet_split_k<<<gbl((long)MTOT*128), 256, 0, stream>>>(F(o_t256b), F(o_net), F(o_inp));

  // ---- flow prior ----
  ic_nchw(IN(IN_PRIOR), 64, 3, 3, 1, 1, 576);
  gemm(BH(o_bufA), W_PF1, F(o_pf1), 1.f, 2);               // gelu
  ic_pm(F(o_pf1), 64, 64, 3, 3, 1, 1, 576);
  gemm(BH(o_bufA), W_PF2, F(o_t16), 1.f, 0);
  coords_init_k<<<gbl(MTOT), 256, 0, stream>>>(F(o_t16), F(o_crd));

  // ---- iterative refinement ----
  for (int it=0; it<6; it++){
    corr_lookup_k<<<gbl((long)MTOT*352), 256, 0, stream>>>(
        F(o_pyr0), F(o_pyr1), F(o_pyr2), F(o_pyr3), F(o_crd), BH(o_bufA));
    flow_k<<<gbl(MTOT), 256, 0, stream>>>(F(o_crd), F(o_flow));

    gemm(BH(o_bufA), W_MC1, F(o_cor1), 1.f, 1);                        // relu
    ic_pm(F(o_cor1), 192, 192, 3, 3, 1, 1, 1728);
    gemm(BH(o_bufA), W_MC2, F(o_cor2), 1.f, 1);
    ic_pm(F(o_flow), 2, 2, 7, 7, 3, 3, 128);
    gemm(BH(o_bufA), W_MF1, F(o_flo1), 1.f, 1);
    ic_pm(F(o_flo1), 64, 64, 3, 3, 1, 1, 576);
    gemm(BH(o_bufA), W_MF2, F(o_flo2), 1.f, 1);
    concat2_k<<<gbl((long)MTOT*160), 256, 0, stream>>>(F(o_cor2),128,128, F(o_flo2),32,32, F(o_cf));
    ic_pm(F(o_cf), 160, 160, 3, 3, 1, 1, 1440);
    gemm(BH(o_bufA), W_MO, F(o_mo), 1.f, 1);
    concat2_k<<<gbl((long)MTOT*128), 256, 0, stream>>>(F(o_mo),128,126, F(o_flow),2,2, F(o_mot));
    concat2_k<<<gbl((long)MTOT*256), 256, 0, stream>>>(F(o_inp),128,128, F(o_mot),128,128, F(o_x));

    // GRU half 1 (1x5, pad (0,2)) then half 2 (5x1, pad (2,0))
    for (int h=0; h<2; h++){
      int KH = h ? 5 : 1, KW = h ? 1 : 5, pH = h ? 2 : 0, pW = h ? 0 : 2;
      int wz = h ? W_GZ2 : W_GZ1, wr = h ? W_GR2 : W_GR1, wq = h ? W_GQ2 : W_GQ1;
      concat2_k<<<gbl((long)MTOT*384), 256, 0, stream>>>(F(o_net),128,128, F(o_x),256,256, F(o_hx));
      ic_pm(F(o_hx), 384, 384, KH, KW, pH, pW, 1920);
      gemm(BH(o_bufA), wz, F(o_zlin), 1.f, 0);
      gemm(BH(o_bufA), wr, F(o_rlin), 1.f, 0);
      rqcat_k<<<gbl((long)MTOT*384), 256, 0, stream>>>(F(o_rlin), F(o_net), F(o_x), F(o_qin));
      ic_pm(F(o_qin), 384, 384, KH, KW, pH, pW, 1920);
      gemm(BH(o_bufA), wq, F(o_qlin), 1.f, 0);
      gru_upd_k<<<gbl((long)MTOT*128), 256, 0, stream>>>(F(o_net), F(o_zlin), F(o_qlin));
    }

    // flow head
    ic_pm(F(o_net), 128, 128, 3, 3, 1, 1, 1152);
    gemm(BH(o_bufA), W_FH1, F(o_t256a), 1.f, 1);
    ic_pm(F(o_t256a), 256, 256, 3, 3, 1, 1, 2304);
    gemm(BH(o_bufA), W_FH2, F(o_t16), 1.f, 0);

    if (it == 5){
      // mask head only needed on the last iteration (only last up_mask is used)
      ic_pm(F(o_net), 128, 128, 3, 3, 1, 1, 1152);
      gemm(BH(o_bufA), W_MH1, F(o_t256a), 1.f, 1);
      ic_pm(F(o_t256a), 256, 256, 1, 1, 0, 0, 256);
      gemm(BH(o_bufA), W_MH2, F(o_mask), 1.f, 0);   // 0.25 folded into upsample softmax
    }

    coords_upd_k<<<gbl(MTOT), 256, 0, stream>>>(F(o_crd), F(o_t16));
  }

  // ---- final flow + convex upsampling ----
  flow_k<<<gbl(MTOT), 256, 0, stream>>>(F(o_crd), F(o_flow));
  upsample_k<<<gbl(2*2*128*128), 256, 0, stream>>>(F(o_flow), F(o_mask), (float*)d_out);
}
